// JKGCN_40638980555140
// MI455X (gfx1250) — matrix operand order repaired and checked by reading, not compiled, not run on hardware
//
#include <hip/hip_runtime.h>
#include <hip/hip_bf16.h>

#define CDIM 64
#define TMAX 8
#define KMAXTOT 288   // max K1+K2 = 64 + 224

typedef __attribute__((ext_vector_type(16))) __bf16 v16bf;
typedef __attribute__((ext_vector_type(8)))  float  v8f;

// ---------- bf16 helpers (round-to-nearest-even) ----------
__device__ __forceinline__ unsigned short bfr(float a) {
  unsigned int u = __float_as_uint(a);
  return (unsigned short)((u + 0x7FFFu + ((u >> 16) & 1u)) >> 16);
}
__device__ __forceinline__ float sigm(float x) { return 1.0f / (1.0f + __expf(-x)); }

// ---------- WMMA fragment loaders ----------
// A fragment from LDS-staged bf16 tile [16][stride].
// ISA 16-bit A 16x32 layout: lane<16 -> k {0..7, 16..23}; lane>=16 -> k {8..15, 24..31}
__device__ __forceinline__ v16bf load_a_lds(const unsigned short* s, int stride,
                                            int lx, int kc, int hi16) {
  const unsigned short* p = s + lx * stride + kc + hi16 * 8;
  union { v16bf v; uint4 q[2]; } r;
  r.q[0] = *(const uint4*)(p);
  r.q[1] = *(const uint4*)(p + 16);
  return r.v;
}
// B: bf16 weight [Nout,K] row-major (torch Linear layout, i.e. Y = A @ W^T).
// B 32x16 layout: lane<16 -> col=lane, k {0..7, 16..23}; lane>=16 -> k {8..15, 24..31}
__device__ __forceinline__ v16bf load_b_frag(const unsigned short* __restrict__ W, int K,
                                             int col, int kc, int hi16) {
  const unsigned short* p = W + (long)col * K + kc + hi16 * 8;
  union { v16bf v; uint4 q[2]; } r;
  r.q[0] = *(const uint4*)(p);
  r.q[1] = *(const uint4*)(p + 16);
  return r.v;
}
// cooperative 16 x Ktot bf16 A-tile stage (A = [A1 | A2] packed per row)
__device__ __forceinline__ void stage_a(unsigned short* As, int m0, int M,
                                        const unsigned short* __restrict__ A1, int K1,
                                        const unsigned short* __restrict__ A2, int K2) {
  const int Ktot = K1 + K2;
  const int rpc = Ktot >> 3;            // 8-elem (16B) chunks per row
  const int chunks = rpc << 4;          // 16 rows
  for (int cidx = threadIdx.x; cidx < chunks; cidx += 256) {
    int r = cidx / rpc;
    int o = (cidx - r * rpc) << 3;
    int grow = m0 + r; if (grow >= M) grow = M - 1;
    const unsigned short* src = (o < K1) ? (A1 + (long)grow * K1 + o)
                                         : (A2 + (long)grow * K2 + (o - K1));
    *(uint4*)(&As[r * Ktot + o]) = *(const uint4*)src;
  }
}

// ---------- generic WMMA linear: Y = A1@W1^T (+ bias) ----------
__global__ void __launch_bounds__(256) wmma_linear(
    const unsigned short* __restrict__ A1, int K1, const unsigned short* __restrict__ W1,
    const float* __restrict__ bias, float* __restrict__ Y,
    int M, int Nout, int ldy) {
  __shared__ unsigned short As[16 * KMAXTOT];
  const int wave = threadIdx.x >> 5;
  const int lane = threadIdx.x & 31;
  const int lx = lane & 15;
  const int hi16 = lane >> 4;
  const int tn = (Nout + 15) >> 4;
  const int m0 = blockIdx.y << 4;

  stage_a(As, m0, M, A1, K1, A1, 0);
  __syncthreads();

  const int ntile = blockIdx.x * 8 + wave;
  if (ntile >= tn) return;
  const int n0 = ntile << 4;
  const int col = n0 + lx;
  const int colc = (col < Nout) ? col : (Nout - 1);

  v8f acc;
  float bv = bias ? bias[colc] : 0.0f;
#pragma unroll
  for (int v = 0; v < 8; ++v) acc[v] = bv;

  for (int kc = 0; kc < K1; kc += 32) {
    v16bf a = load_a_lds(As, K1, lx, kc, hi16);
    v16bf b = load_b_frag(W1, K1, colc, kc, hi16);
    acc = __builtin_amdgcn_wmma_f32_16x16x32_bf16(false, a, false, b, (short)0, acc, false, false);
  }
  if (col < Nout) {
#pragma unroll
    for (int v = 0; v < 8; ++v) {
      int rr = m0 + v + (hi16 << 3);
      if (rr < M) Y[(long)rr * ldy + col] = acc[v];
    }
  }
}

// ---------- fused LSTM step ----------
// Gates = xs_t @ Wih^T + h_prev @ Whh^T + bias, then nonlinearity + c/h update, all in one kernel.
// Each wave owns hidden-unit column tile j0 and computes 4 accumulators (gates i,f,g,o at
// columns g*hd + j0), reusing one LDS A-fragment across 4 WMMAs per K-chunk.
// h is double-buffered by the caller (reads hprev_bf as A2, writes hout/hbf_out).
__global__ void __launch_bounds__(256) wmma_lstm_step(
    const unsigned short* __restrict__ Axs, int K1, const unsigned short* __restrict__ Wih,
    const unsigned short* __restrict__ Ahp, int K2, const unsigned short* __restrict__ Whh,
    const float* __restrict__ bias,            // [4*hd]  (bih + bhh)
    float* __restrict__ c,                     // [M, hd] in/out
    float* __restrict__ hout,                  // [M, hd] f32 (for attention dot)
    unsigned short* __restrict__ hbf_out,      // [M, hd] bf16 (next step A operand)
    int M, int hd) {
  __shared__ unsigned short As[16 * KMAXTOT];
  const int wave = threadIdx.x >> 5;
  const int lane = threadIdx.x & 31;
  const int lx = lane & 15;
  const int hi16 = lane >> 4;
  const int m0 = blockIdx.y << 4;
  const int Ktot = K1 + K2;
  const int jt = hd >> 4;                      // hidden-unit column tiles

  stage_a(As, m0, M, Axs, K1, Ahp, K2);
  __syncthreads();

  const int jtile = blockIdx.x * 8 + wave;
  if (jtile >= jt) return;
  const int j0 = jtile << 4;
  const int colh = j0 + lx;                    // hidden unit index, always < hd

  v8f ai, af, ag, ao;
  {
    float b0 = bias[colh], b1 = bias[hd + colh], b2 = bias[2 * hd + colh], b3 = bias[3 * hd + colh];
#pragma unroll
    for (int v = 0; v < 8; ++v) { ai[v] = b0; af[v] = b1; ag[v] = b2; ao[v] = b3; }
  }

  for (int kc = 0; kc < K1; kc += 32) {
    v16bf a  = load_a_lds(As, Ktot, lx, kc, hi16);
    v16bf b0 = load_b_frag(Wih, K1, colh, kc, hi16);
    v16bf b1 = load_b_frag(Wih, K1, hd + colh, kc, hi16);
    v16bf b2 = load_b_frag(Wih, K1, 2 * hd + colh, kc, hi16);
    v16bf b3 = load_b_frag(Wih, K1, 3 * hd + colh, kc, hi16);
    ai = __builtin_amdgcn_wmma_f32_16x16x32_bf16(false, a, false, b0, (short)0, ai, false, false);
    af = __builtin_amdgcn_wmma_f32_16x16x32_bf16(false, a, false, b1, (short)0, af, false, false);
    ag = __builtin_amdgcn_wmma_f32_16x16x32_bf16(false, a, false, b2, (short)0, ag, false, false);
    ao = __builtin_amdgcn_wmma_f32_16x16x32_bf16(false, a, false, b3, (short)0, ao, false, false);
  }
  for (int kc = 0; kc < K2; kc += 32) {
    v16bf a  = load_a_lds(As, Ktot, lx, K1 + kc, hi16);
    v16bf b0 = load_b_frag(Whh, K2, colh, kc, hi16);
    v16bf b1 = load_b_frag(Whh, K2, hd + colh, kc, hi16);
    v16bf b2 = load_b_frag(Whh, K2, 2 * hd + colh, kc, hi16);
    v16bf b3 = load_b_frag(Whh, K2, 3 * hd + colh, kc, hi16);
    ai = __builtin_amdgcn_wmma_f32_16x16x32_bf16(false, a, false, b0, (short)0, ai, false, false);
    af = __builtin_amdgcn_wmma_f32_16x16x32_bf16(false, a, false, b1, (short)0, af, false, false);
    ag = __builtin_amdgcn_wmma_f32_16x16x32_bf16(false, a, false, b2, (short)0, ag, false, false);
    ao = __builtin_amdgcn_wmma_f32_16x16x32_bf16(false, a, false, b3, (short)0, ao, false, false);
  }

  // fused pointwise epilogue: c' = sig(f)*c + sig(i)*tanh(g); h = sig(o)*tanh(c')
#pragma unroll
  for (int v = 0; v < 8; ++v) {
    int row = m0 + v + (hi16 << 3);
    if (row < M) {
      long idx = (long)row * hd + colh;
      float cp = c[idx];
      float cn = sigm(af[v]) * cp + sigm(ai[v]) * tanhf(ag[v]);
      c[idx] = cn;
      float hn = sigm(ao[v]) * tanhf(cn);
      hout[idx] = hn;
      hbf_out[idx] = bfr(hn);
    }
  }
}

// ---------- small utility kernels ----------
__global__ void fill_f32(float* __restrict__ p, float v, long n) {
  long i = (long)blockIdx.x * blockDim.x + threadIdx.x;
  if (i < n) p[i] = v;
}
__global__ void cvt_bf16(const float* __restrict__ s, unsigned short* __restrict__ d, long n) {
  long i = (long)blockIdx.x * blockDim.x + threadIdx.x;
  if (i < n) d[i] = bfr(s[i]);
}
__global__ void transpose_cvt64(const float* __restrict__ s, unsigned short* __restrict__ d) {
  int i = blockIdx.x * blockDim.x + threadIdx.x;  // 4096 elems
  if (i < 64 * 64) { int o = i >> 6, k = i & 63; d[i] = bfr(s[k * 64 + o]); }
}
__global__ void vadd(const float* __restrict__ a, const float* __restrict__ b,
                     float* __restrict__ o, int n) {
  int i = blockIdx.x * blockDim.x + threadIdx.x;
  if (i < n) o[i] = a[i] + b[i];
}
__global__ void deg_count(const int* __restrict__ dst, float* __restrict__ deg, int E) {
  int e = blockIdx.x * blockDim.x + threadIdx.x;
  if (e < E) atomicAdd(&deg[dst[e]], 1.0f);
}
__global__ void dinv_k(float* __restrict__ d, int N) {
  int n = blockIdx.x * blockDim.x + threadIdx.x;
  if (n < N) { float x = d[n]; d[n] = (x > 0.f) ? rsqrtf(x) : 0.f; }
}
__global__ void norm_k(const int* __restrict__ s, const int* __restrict__ dd,
                       const float* __restrict__ dinv, float* __restrict__ nrm, int E, int N) {
  int e = blockIdx.x * blockDim.x + threadIdx.x;
  if (e < E + N) {
    int a = (e < E) ? s[e] : (e - E);
    int b = (e < E) ? dd[e] : (e - E);
    nrm[e] = dinv[a] * dinv[b];
  }
}
__global__ void alpha_dot(const float* __restrict__ h, const float* __restrict__ w,
                          float* __restrict__ alpha, int N, int hd, int t) {
  int n = blockIdx.x * blockDim.x + threadIdx.x;
  if (n >= N) return;
  const float* p = h + (long)n * hd;
  float s = 0.f;
  for (int j = 0; j < hd; ++j) s += p[j] * w[j];
  alpha[n * TMAX + t] += s;
}
__global__ void softmax_T(float* __restrict__ alpha, const float* __restrict__ attb,
                          int N, int T) {
  int n = blockIdx.x * blockDim.x + threadIdx.x;
  if (n >= N) return;
  float ab = *attb;
  float v[TMAX]; float m = -1e30f;
  for (int t = 0; t < T; ++t) { v[t] = alpha[n * TMAX + t] + ab; m = fmaxf(m, v[t]); }
  float s = 0.f;
  for (int t = 0; t < T; ++t) { v[t] = __expf(v[t] - m); s += v[t]; }
  float inv = 1.f / s;
  for (int t = 0; t < T; ++t) alpha[n * TMAX + t] = v[t] * inv;
}
// attention-weighted sum of layer stack -> bf16 (only ever consumed as a WMMA A-operand)
__global__ void jk_weighted(const float* __restrict__ alpha, const float* __restrict__ xs,
                            unsigned short* __restrict__ out, int N, int T) {
  long i = (long)blockIdx.x * blockDim.x + threadIdx.x;
  if (i >= (long)N * CDIM) return;
  int n = (int)(i >> 6);
  float s = 0.f;
  for (int t = 0; t < T; ++t) s += alpha[n * TMAX + t] * xs[(long)t * N * CDIM + i];
  out[i] = bfr(s);
}
__global__ void scatter_edges(const float* __restrict__ h, const float* __restrict__ nrm,
                              const int* __restrict__ e0, const int* __restrict__ e1,
                              float* __restrict__ acc, int E, int N) {
  long i = (long)blockIdx.x * blockDim.x + threadIdx.x;
  long total = (long)(E + N) * CDIM;
  if (i >= total) return;
  int e = (int)(i >> 6);
  int c = (int)(i & 63);
  int s = (e < E) ? e0[e] : (e - E);
  int d = (e < E) ? e1[e] : (e - E);
  atomicAdd(&acc[(long)d * CDIM + c], h[(long)s * CDIM + c] * nrm[e]);
}
// bias + relu -> f32 layer output (for jk_weighted) and bf16 copy (for WMMA A)
__global__ void bias_relu(const float* __restrict__ acc, const float* __restrict__ b,
                          float* __restrict__ out, unsigned short* __restrict__ outbf, int N) {
  long i = (long)blockIdx.x * blockDim.x + threadIdx.x;
  if (i >= (long)N * CDIM) return;
  int c = (int)(i & 63);
  float v = acc[i] + b[c];
  v = v > 0.f ? v : 0.f;
  out[i] = v;
  outbf[i] = bfr(v);
}
__global__ void log_softmax40(const float* __restrict__ lg, float* __restrict__ out, int N) {
  int n = blockIdx.x * blockDim.x + threadIdx.x;
  if (n >= N) return;
  const float* p = lg + (long)n * 40;
  float m = -1e30f;
  for (int c = 0; c < 40; ++c) m = fmaxf(m, p[c]);
  float s = 0.f;
  for (int c = 0; c < 40; ++c) s += __expf(p[c] - m);
  float l = logf(s) + m;
  for (int c = 0; c < 40; ++c) out[(long)n * 40 + c] = p[c] - l;
}

extern "C" void kernel_launch(void* const* d_in, const int* in_sizes, int n_in,
                              void* d_out, int out_size, void* d_ws, size_t ws_size,
                              hipStream_t stream) {
  const int Cc = 64, L = 6;
  const int N = in_sizes[0] / Cc;
  const int E = in_sizes[n_in - 1] / 2;

  const float* x     = (const float*)d_in[0];
  const float* gcn_w = (const float*)d_in[1];
  const float* gcn_b = (const float*)d_in[2];
  // d_in[3 .. 62]: jk_params[0..5] x {Wih_f,Whh_f,bih_f,bhh_f,Wih_b,Whh_b,bih_b,bhh_b,att_w,att_b}
  // d_in[63 .. 72]: jk_final (same 10)
  const float* lin_w = (const float*)d_in[3 + 7 * 10];
  const float* lin_b = (const float*)d_in[3 + 7 * 10 + 1];
  const int*   ei    = (const int*)d_in[3 + 7 * 10 + 2];
  const int* e0 = ei;
  const int* e1 = ei + E;

  size_t off = 0;
  auto alloc = [&](size_t bytes) -> void* {
    void* p = (char*)d_ws + off;
    off = (off + bytes + 255) & ~(size_t)255;
    return p;
  };
  float*          xs    = (float*)alloc((size_t)7 * N * Cc * 4);
  unsigned short* xs_bf = (unsigned short*)alloc((size_t)7 * N * Cc * 2);
  float*          hq    = (float*)alloc((size_t)N * 224 * 4);
  unsigned short* hbfA  = (unsigned short*)alloc((size_t)N * 224 * 2);
  unsigned short* hbfB  = (unsigned short*)alloc((size_t)N * 224 * 2);
  float*          cq    = (float*)alloc((size_t)N * 224 * 4);
  float*          alpha = (float*)alloc((size_t)N * TMAX * 4);
  unsigned short* hjk_bf= (unsigned short*)alloc((size_t)N * Cc * 2);
  float*          hlin  = (float*)alloc((size_t)N * Cc * 4);
  float*          acc   = (float*)alloc((size_t)N * Cc * 4);
  float*          dinv  = (float*)alloc((size_t)N * 4);
  float*          nrm   = (float*)alloc((size_t)(E + N) * 4);
  float*          biasF = (float*)alloc(896 * 4);
  float*          biasB = (float*)alloc(896 * 4);
  unsigned short *wihF[7], *whhF[7], *wihB[7], *whhB[7];
  for (int s = 0; s < 7; ++s) {
    int T = s + 1, h = 32 * T, g4 = 4 * h;
    wihF[s] = (unsigned short*)alloc((size_t)g4 * Cc * 2);
    whhF[s] = (unsigned short*)alloc((size_t)g4 * h * 2);
    wihB[s] = (unsigned short*)alloc((size_t)g4 * Cc * 2);
    whhB[s] = (unsigned short*)alloc((size_t)g4 * h * 2);
  }
  unsigned short* gcnT  = (unsigned short*)alloc((size_t)L * Cc * Cc * 2);
  unsigned short* linBf = (unsigned short*)alloc((size_t)40 * Cc * 2);
  (void)ws_size; (void)out_size;

  auto blk = [](long n) { return (int)((n + 255) / 256); };

  // ---- weight conversion (bf16; gcn_w transposed so GEMM applies W^T form) ----
  for (int s = 0; s < 7; ++s) {
    int T = s + 1, h = 32 * T, g4 = 4 * h;
    int base = (s < 6) ? (3 + s * 10) : 63;
    cvt_bf16<<<blk((long)g4 * Cc), 256, 0, stream>>>((const float*)d_in[base + 0], wihF[s], (long)g4 * Cc);
    cvt_bf16<<<blk((long)g4 * h),  256, 0, stream>>>((const float*)d_in[base + 1], whhF[s], (long)g4 * h);
    cvt_bf16<<<blk((long)g4 * Cc), 256, 0, stream>>>((const float*)d_in[base + 4], wihB[s], (long)g4 * Cc);
    cvt_bf16<<<blk((long)g4 * h),  256, 0, stream>>>((const float*)d_in[base + 5], whhB[s], (long)g4 * h);
  }
  for (int i = 0; i < L; ++i)
    transpose_cvt64<<<16, 256, 0, stream>>>(gcn_w + (size_t)i * Cc * Cc, gcnT + (size_t)i * Cc * Cc);
  cvt_bf16<<<blk(40 * Cc), 256, 0, stream>>>(lin_w, linBf, 40 * Cc);

  // ---- degree / symmetric norm (self loops: init deg = 1) ----
  fill_f32<<<blk(N), 256, 0, stream>>>(dinv, 1.0f, N);
  deg_count<<<blk(E), 256, 0, stream>>>(e1, dinv, E);
  dinv_k<<<blk(N), 256, 0, stream>>>(dinv, N);
  norm_k<<<blk(E + N), 256, 0, stream>>>(e0, e1, dinv, nrm, E, N);

  // xs[0] = x (f32 for attention-weighted sums, bf16 for WMMA A-operand)
  hipMemcpyAsync(xs, x, (size_t)N * Cc * 4, hipMemcpyDeviceToDevice, stream);
  cvt_bf16<<<blk((long)N * Cc), 256, 0, stream>>>(x, xs_bf, (long)N * Cc);

  auto linear = [&](const unsigned short* A1, int K1, const unsigned short* W1,
                    const float* bias, float* Y, int M, int Nout, int ldy) {
    int tm = (M + 15) / 16, tn = (Nout + 15) / 16;
    dim3 grid((tn + 7) / 8, tm);
    wmma_linear<<<grid, 256, 0, stream>>>(A1, K1, W1, bias, Y, M, Nout, ldy);
  };

  auto run_jk = [&](int s, unsigned short* outv) {
    int T = s + 1, h = 32 * T, g4 = 4 * h;
    int base = (s < 6) ? (3 + s * 10) : 63;
    const float* bihF = (const float*)d_in[base + 2];
    const float* bhhF = (const float*)d_in[base + 3];
    const float* bihB = (const float*)d_in[base + 6];
    const float* bhhB = (const float*)d_in[base + 7];
    const float* attw = (const float*)d_in[base + 8];
    const float* attb = (const float*)d_in[base + 9];
    vadd<<<blk(g4), 256, 0, stream>>>(bihF, bhhF, biasF, g4);
    vadd<<<blk(g4), 256, 0, stream>>>(bihB, bhhB, biasB, g4);
    fill_f32<<<blk((long)N * TMAX), 256, 0, stream>>>(alpha, 0.f, (long)N * TMAX);
    int tm = (N + 15) / 16, jt = h / 16;
    dim3 lgrid((jt + 7) / 8, tm);
    for (int dir = 0; dir < 2; ++dir) {
      const unsigned short* wih = dir ? wihB[s] : wihF[s];
      const unsigned short* whh = dir ? whhB[s] : whhF[s];
      const float* bb = dir ? biasB : biasF;
      const float* aw = attw + (dir ? h : 0);   // out = concat([hf, hb])
      // h0 = c0 = 0 (bf16 zero == 0x0000 bit pattern)
      fill_f32<<<blk((long)N * h / 2), 256, 0, stream>>>((float*)hbfA, 0.f, (long)N * h / 2);
      fill_f32<<<blk((long)N * h), 256, 0, stream>>>(cq, 0.f, (long)N * h);
      unsigned short* hin = hbfA;
      unsigned short* hout = hbfB;
      for (int st = 0; st < T; ++st) {
        int t = dir ? (T - 1 - st) : st;
        // fused: gates GEMM (xs_t@Wih^T + h@Whh^T + bias) + LSTM pointwise + h/c update
        wmma_lstm_step<<<lgrid, 256, 0, stream>>>(
            xs_bf + (size_t)t * N * Cc, Cc, wih, hin, h, whh, bb, cq, hq, hout, N, h);
        { unsigned short* tmp = hin; hin = hout; hout = tmp; }
        alpha_dot<<<blk(N), 256, 0, stream>>>(hq, aw, alpha, N, h, t);
      }
    }
    softmax_T<<<blk(N), 256, 0, stream>>>(alpha, attb, N, T);
    jk_weighted<<<blk((long)N * Cc), 256, 0, stream>>>(alpha, xs, outv, N, T);
  };

  for (int i = 0; i < L; ++i) {
    run_jk(i, hjk_bf);
    linear(hjk_bf, Cc, gcnT + (size_t)i * Cc * Cc, nullptr, hlin, N, Cc, Cc);
    fill_f32<<<blk((long)N * Cc), 256, 0, stream>>>(acc, 0.f, (long)N * Cc);
    scatter_edges<<<blk((long)(E + N) * Cc), 256, 0, stream>>>(hlin, nrm, e0, e1, acc, E, N);
    bias_relu<<<blk((long)N * Cc), 256, 0, stream>>>(acc, gcn_b + (size_t)i * Cc,
                                                     xs + (size_t)(i + 1) * N * Cc,
                                                     xs_bf + (size_t)(i + 1) * N * Cc, N);
  }
  run_jk(6, hjk_bf);
  linear(hjk_bf, Cc, linBf, lin_b, hlin, N, 40, 40);
  log_softmax40<<<blk(N), 256, 0, stream>>>(hlin, (float*)d_out, N);
}
